// Attention_75007308857927
// MI455X (gfx1250) — compile-verified, matrix operand order checked by reading, not run on hardware
//
#include <hip/hip_runtime.h>
#include <hip/hip_bf16.h>

#define B_ 4
#define S_ 2048
#define D_ 1024
#define BS_ (B_ * S_)
#define BSD_ ((size_t)BS_ * D_)

typedef __attribute__((ext_vector_type(16))) _Float16 v16h;
typedef __attribute__((ext_vector_type(4)))  _Float16 v4h;
typedef __attribute__((ext_vector_type(8)))  float    v8f;
typedef __attribute__((ext_vector_type(4)))  float    vf4;

// Load a 16-bit A/B WMMA fragment for 16x16x32 ops.
// Per ISA layout: lane L (L<16) holds K=0..7 (v0-3) and K=16..23 (v4-7);
// lane L+16 holds K=8..15 and K=24..31.  base points at (row-or-col start + k0).
__device__ __forceinline__ v16h load_frag(const _Float16* base, int hi8) {
  union { v16h v; vf4 f[2]; } u;
  u.f[0] = *(const vf4*)(base + hi8);
  u.f[1] = *(const vf4*)(base + 16 + hi8);
  return u.v;
}

__device__ __forceinline__ v8f wmma_f16(v16h a, v16h b, v8f c) {
  return __builtin_amdgcn_wmma_f32_16x16x32_f16(false, a, false, b, (short)0, c,
                                                false, false);
}

__device__ __forceinline__ v8f zero8() {
  v8f z = {0.f, 0.f, 0.f, 0.f, 0.f, 0.f, 0.f, 0.f};
  return z;
}

// ---------------- fp32 -> f16 conversion ----------------
__global__ __launch_bounds__(256) void k_cvt(const float* __restrict__ in,
                                             _Float16* __restrict__ out, int n) {
  int i = blockIdx.x * 256 + threadIdx.x;
  if (i < n) out[i] = (_Float16)in[i];
}

// ---------------- QKV projection: Y = Xh @ W^T  (W row-major [e][d]) -------
// grid (BS/16, 3), block 256 (8 waves). Workgroup: 16 rows x 1024 cols.
__global__ __launch_bounds__(256) void k_gemm_qkv(const _Float16* __restrict__ Xh,
                                                  const _Float16* __restrict__ Wh3,
                                                  _Float16* __restrict__ QKVh) {
  const int rb  = blockIdx.x;
  const int y   = blockIdx.y;
  const int tid = threadIdx.x;
  const int w   = tid >> 5;
  const int lane = tid & 31;
  const int ln  = lane & 15;
  const int hi8 = (lane >> 4) << 3;

  const _Float16* W = Wh3 + (size_t)y * D_ * D_;
  _Float16*       O = QKVh + (size_t)y * BSD_;
  const int row0 = rb * 16;
  const int c0   = w * 128;

  v8f acc[8];
#pragma unroll
  for (int j = 0; j < 8; ++j) acc[j] = zero8();

  const _Float16* arow = Xh + (size_t)(row0 + ln) * D_;
  for (int d0 = 0; d0 < D_; d0 += 32) {
    v16h a = load_frag(arow + d0, hi8);
#pragma unroll
    for (int j = 0; j < 8; ++j) {
      v16h b = load_frag(W + (size_t)(c0 + j * 16 + ln) * D_ + d0, hi8);
      acc[j] = wmma_f16(a, b, acc[j]);
    }
  }
  // C/D layout: vgpr r, lanes 0-15 -> M=r, lanes 16-31 -> M=r+8 (== r+hi8)
#pragma unroll
  for (int j = 0; j < 8; ++j)
#pragma unroll
    for (int r = 0; r < 8; ++r)
      O[(size_t)(row0 + r + hi8) * D_ + c0 + j * 16 + ln] = (_Float16)acc[j][r];
}

// ---------------- row L2-normalize Q and K in place ----------------
// grid (BS, 2), block 256; each thread owns 4 halfs of the 1024-wide row.
__global__ __launch_bounds__(256) void k_normalize(_Float16* __restrict__ QKVh) {
  __shared__ float red[256];
  const int row = blockIdx.x;
  const int y   = blockIdx.y;   // 0 = Q, 1 = K
  const int tid = threadIdx.x;
  _Float16* p = QKVh + (size_t)y * BSD_ + (size_t)row * D_ + tid * 4;
  v4h v = *(v4h*)p;
  float s = 0.f;
#pragma unroll
  for (int q = 0; q < 4; ++q) { float f = (float)v[q]; s += f * f; }
  red[tid] = s;
  __syncthreads();
  for (int st = 128; st > 0; st >>= 1) {
    if (tid < st) red[tid] += red[tid + st];
    __syncthreads();
  }
  float inv = rsqrtf(red[0]);
#pragma unroll
  for (int q = 0; q < 4; ++q) v[q] = (_Float16)((float)v[q] * inv);
  *(v4h*)p = v;
}

// ---------------- V transpose per batch: Vh[s][d] -> Vt[d][s] ----------------
// grid (S/32, D/32, B), block 256, 32x32 LDS tile.
__global__ __launch_bounds__(256) void k_transpose(const _Float16* __restrict__ Vh,
                                                   _Float16* __restrict__ Vt) {
  __shared__ _Float16 tile[32][33];
  const int b  = blockIdx.z;
  const int s0 = blockIdx.x * 32;
  const int d0 = blockIdx.y * 32;
  const int t  = threadIdx.x;
  const int i  = t >> 3, j4 = (t & 7) << 2;
  const _Float16* src = Vh + (size_t)b * S_ * D_ + (size_t)(s0 + i) * D_ + d0 + j4;
#pragma unroll
  for (int q = 0; q < 4; ++q) tile[i][j4 + q] = src[q];
  __syncthreads();
  const int j = t >> 3, i4 = (t & 7) << 2;
  _Float16* dst = Vt + (size_t)b * D_ * S_ + (size_t)(d0 + j) * S_ + s0 + i4;
#pragma unroll
  for (int q = 0; q < 4; ++q) dst[q] = tile[i4 + q][j];
}

// ---------------- attention: out = (Qn @ Kn^T * m) @ V ----------------
// Reference semantics: scores are MULTIPLIED by the mask, where mask==1 at
// MASKED positions (future OR padding). keep = (k > q) || (pad[k]==0).
// grid (BS/16), block 256 (8 waves). Workgroup: 16 queries x 1024 cols.
// Wave w: GEMM-1 partial for score tile (w&1) over D slice (w>>1)*256;
//         GEMM-2 accumulation for output columns [w*128, w*128+128).
__global__ __launch_bounds__(256) void k_attn(const _Float16* __restrict__ Qh,
                                              const _Float16* __restrict__ Kh,
                                              const _Float16* __restrict__ Vt,
                                              const int* __restrict__ masks,
                                              float* __restrict__ out) {
  __shared__ float sp[2][8 * 256];   // double-buffered partial score tiles
  __shared__ int   smask[S_];
  const int blk = blockIdx.x;
  const int b   = blk / (S_ / 16);
  const int q0  = (blk % (S_ / 16)) * 16;
  const int tid = threadIdx.x;
  const int w   = tid >> 5;
  const int lane = tid & 31;
  const int ln  = lane & 15;
  const int hi8 = (lane >> 4) << 3;

  for (int i = tid; i < S_; i += 256) smask[i] = masks[b * S_ + i];

  v8f acc[8];
#pragma unroll
  for (int j = 0; j < 8; ++j) acc[j] = zero8();

  const int tile = w & 1;          // which 16-key score tile this wave partials
  const int ds0  = (w >> 1) << 8;  // 256-wide D slice for the score reduction
  const size_t rowQ = (size_t)(b * S_ + q0 + ln) * D_;
  const int q_me = q0 + ln;
  const int c0   = w * 128;        // output column slice
  const size_t vbase = (size_t)b * D_ * S_;

  // Loop-invariant Q fragments for this wave's D slice (16 x 256).
  v16h qf[8];
#pragma unroll
  for (int s = 0; s < 8; ++s) qf[s] = load_frag(Qh + rowQ + ds0 + s * 32, hi8);

  __syncthreads();  // smask ready

  for (int kb = 0; kb < S_ / 32; ++kb) {
    const int k0 = kb * 32;
    float* spp = sp[kb & 1];

    // --- GEMM 1 partial: scores[q][k] over this wave's D slice ---
    v8f sc = zero8();
    const size_t rowK = (size_t)(b * S_ + k0 + tile * 16 + ln) * D_;
#pragma unroll
    for (int s = 0; s < 8; ++s) {
      v16h kf = load_frag(Kh + rowK + ds0 + s * 32, hi8);
      sc = wmma_f16(qf[s], kf, sc);
    }
    // park partial tile in LDS as [wave][m][n]
#pragma unroll
    for (int r = 0; r < 8; ++r) spp[w * 256 + (r + hi8) * 16 + ln] = sc[r];
    __syncthreads();

    // --- reduce 4 D-slices, apply mask, build f16 A-fragment (16x32) ---
    union { v16h v; _Float16 h[16]; } sa;
#pragma unroll
    for (int j = 0; j < 16; ++j) {
      const int kidx = (j < 8) ? (hi8 + j) : (16 + hi8 + j - 8);
      const int t2 = kidx >> 4, n2 = kidx & 15;
      const int base = ln * 16 + n2;          // m = ln (query row of A-frag)
      float s4 = spp[(t2    ) * 256 + base] + spp[(t2 + 2) * 256 + base] +
                 spp[(t2 + 4) * 256 + base] + spp[(t2 + 6) * 256 + base];
      const int kg = k0 + kidx;
      const bool keep = (kg > q_me) || (smask[kg] == 0);
      sa.h[j] = keep ? (_Float16)s4 : (_Float16)0.f;
    }

    // --- GEMM 2: acc += scores(16x32) @ V(32 x 128 cols of this wave) ---
#pragma unroll
    for (int jt = 0; jt < 8; ++jt) {
      v16h bv = load_frag(Vt + vbase + (size_t)(c0 + jt * 16 + ln) * S_ + k0, hi8);
      acc[jt] = wmma_f16(sa.v, bv, acc[jt]);
    }
    // No trailing barrier: next iteration writes the other sp buffer, and a
    // wave cannot reach iteration kb+2 (same buffer) without every wave first
    // passing iteration kb+1's barrier, which postdates all reads above.
  }

  // --- write fp32 output ---
#pragma unroll
  for (int jt = 0; jt < 8; ++jt)
#pragma unroll
    for (int r = 0; r < 8; ++r)
      out[(size_t)(b * S_ + q0 + r + hi8) * D_ + c0 + jt * 16 + ln] = acc[jt][r];
}

extern "C" void kernel_launch(void* const* d_in, const int* in_sizes, int n_in,
                              void* d_out, int out_size, void* d_ws, size_t ws_size,
                              hipStream_t stream) {
  const float* X     = (const float*)d_in[0];
  const int*   masks = (const int*)d_in[1];
  const float* Wq    = (const float*)d_in[2];
  const float* Wk    = (const float*)d_in[3];
  const float* Wv    = (const float*)d_in[4];
  float* out = (float*)d_out;

  char* ws = (char*)d_ws;
  _Float16* Xh   = (_Float16*)ws;                              // 16 MB
  _Float16* Wh   = (_Float16*)(ws + ((size_t)16 << 20));       //  6 MB (3 x D*D)
  _Float16* QKVh = (_Float16*)(ws + ((size_t)22 << 20));       // 48 MB (Q,K,V)
  _Float16* Vt   = (_Float16*)(ws + ((size_t)70 << 20));       // 16 MB

  const int nX = (int)BSD_;
  const int nW = D_ * D_;
  k_cvt<<<(nX + 255) / 256, 256, 0, stream>>>(X, Xh, nX);
  k_cvt<<<(nW + 255) / 256, 256, 0, stream>>>(Wq, Wh, nW);
  k_cvt<<<(nW + 255) / 256, 256, 0, stream>>>(Wk, Wh + (size_t)nW, nW);
  k_cvt<<<(nW + 255) / 256, 256, 0, stream>>>(Wv, Wh + 2 * (size_t)nW, nW);

  k_gemm_qkv<<<dim3(BS_ / 16, 3), 256, 0, stream>>>(Xh, Wh, QKVh);
  k_normalize<<<dim3(BS_, 2), 256, 0, stream>>>(QKVh);
  k_transpose<<<dim3(S_ / 32, D_ / 32, B_), 256, 0, stream>>>(QKVh + 2 * BSD_, Vt);
  k_attn<<<BS_ / 16, 256, 0, stream>>>(QKVh, QKVh + BSD_, Vt, masks, out);
}
